// HierarchicalNAM_36129264894761
// MI455X (gfx1250) — matrix operand order repaired and checked by reading, not compiled, not run on hardware
//
#include <hip/hip_runtime.h>

// ---------------------------------------------------------------------------
// HierarchicalNAM on gfx1250 (MI455X, wave32, WMMA)
//
//   out[b] = sum_f [ 0.7 * mlp_brand_f(x[b,f]) + 0.3 * mlp_sku_f(x[b,f]) ]
//   each mlp: 1 -> 64 -> 32 -> 1 with relu on the two hidden layers.
//
// Layer 2 (64x32 GEMM) runs on v_wmma_f32_16x16x32_f16 (2 K-steps x 2 N-tiles
// per pair per 16-row tile). Layer 1 is produced lane-locally in WMMA
// A-fragment layout via packed f16 FMA; layer 3 is an f32 reduction across
// the 16 lanes carrying the N dimension. HIER weights folded into w3 / bias.
//
// v2: model loop rolled -> <256 VGPRs, no s_set_vgpr_msb in the hot loop.
// v3: f32 relu via v_med3 (fmed3(x,0,+inf)) instead of fmaxf, eliminating the
//     v_max_f32 v,v,v NaN-canonicalize the compiler pairs with maxnum.
// ---------------------------------------------------------------------------

typedef __attribute__((ext_vector_type(16))) _Float16 v16h;
typedef __attribute__((ext_vector_type(8)))  float    v8f;

#define BATCH   131072
#define NF      32
#define NPAIR   64          // 32 features x {brand, sku}
#define HIERW   0.7f

// workspace byte offsets (all 32B aligned)
#define OFF_W2  0
#define SZ_W2   (NPAIR * 4 * 1024)          // 64 pairs x (2 ks x 2 nt) frags x 1KB
#define OFF_W1  (OFF_W2 + SZ_W2)
#define SZ_W1   (NPAIR * 2 * 2 * 16 * 2)    // [pair][ks][half][16] f16
#define OFF_B1  (OFF_W1 + SZ_W1)
#define SZ_B1   SZ_W1
#define OFF_B2  (OFF_B1 + SZ_B1)
#define SZ_B2   (NPAIR * 32 * 4)            // f32 [pair][32]
#define OFF_W3  (OFF_B2 + SZ_B2)
#define SZ_W3   SZ_B2
#define OFF_SB  (OFF_W3 + SZ_W3)            // one f32: sum_p hw[p]*b3[p]

__device__ __forceinline__ float relu_f32(float v) {
    // single v_med3: clamp(v, 0, +inf); avoids maxnum canonicalize pair
    return __builtin_amdgcn_fmed3f(v, 0.0f, __builtin_inff());
}

// ---------------------------------------------------------------------------
// Pack kernel: rearrange f32 weights into WMMA fragment order (f16) once.
//
// B-fragment (32x16 f16, K x N): lane L, element j (of v16h):
//     k_local = j + 16*(L>=16),  n = nt*16 + (L&15)
// A-side w1/b1 fragment K-set for 16x32 f16 A tile: lane-half h, element j:
//     K = j + 16*h   (h==0 -> {0..15}, h==1 -> {16..31}), matching B
// ---------------------------------------------------------------------------
__global__ void nam_pack(const float* __restrict__ bW1, const float* __restrict__ bb1,
                         const float* __restrict__ bW2, const float* __restrict__ bb2,
                         const float* __restrict__ bW3, const float* __restrict__ bb3,
                         const float* __restrict__ sW1, const float* __restrict__ sb1,
                         const float* __restrict__ sW2, const float* __restrict__ sb2,
                         const float* __restrict__ sW3, const float* __restrict__ sb3,
                         char* __restrict__ ws) {
    const int p = blockIdx.x;           // pair id
    const int f = p >> 1;
    const int model = p & 1;            // 0 = brand, 1 = sku
    const float* W1 = model ? sW1 : bW1;
    const float* B1 = model ? sb1 : bb1;
    const float* W2 = model ? sW2 : bW2;
    const float* B2 = model ? sb2 : bb2;
    const float* W3 = model ? sW3 : bW3;
    const float  hw = model ? (1.0f - HIERW) : HIERW;

    _Float16* w2pk = (_Float16*)(ws + OFF_W2);
    _Float16* w1pk = (_Float16*)(ws + OFF_W1);
    _Float16* b1pk = (_Float16*)(ws + OFF_B1);
    float*    b2s  = (float*)(ws + OFF_B2);
    float*    w3s  = (float*)(ws + OFF_W3);

    const int tid = threadIdx.x;

    // W2 fragments: 2048 f16 per pair
    for (int idx = tid; idx < 2048; idx += 256) {
        const int j    = idx & 15;
        const int lane = (idx >> 4) & 31;
        const int nt   = (idx >> 9) & 1;
        const int ks   = idx >> 10;
        const int k    = ks * 32 + j + ((lane >= 16) ? 16 : 0);
        const int n    = nt * 16 + (lane & 15);
        w2pk[((p * 2 + ks) * 2 + nt) * 512 + lane * 16 + j] =
            (_Float16)W2[f * 2048 + k * 32 + n];
    }

    // w1 / b1 A-side fragments: 64 f16 per pair per array
    if (tid < 64) {
        const int j  = tid & 15;
        const int h  = (tid >> 4) & 1;
        const int ks = tid >> 5;
        const int kl = j + 16 * h;
        const int k  = ks * 32 + kl;
        w1pk[((p * 2 + ks) * 2 + h) * 16 + j] = (_Float16)W1[f * 64 + k];
        b1pk[((p * 2 + ks) * 2 + h) * 16 + j] = (_Float16)B1[f * 64 + k];
    }

    // b2 (raw), w3 (hier-weighted)
    if (tid < 32) {
        b2s[p * 32 + tid] = B2[f * 32 + tid];
        w3s[p * 32 + tid] = hw * W3[f * 32 + tid];
    }

    // global bias term: sum over all pairs of hw * b3 (same for every row)
    if (p == 0 && tid == 0) {
        float s = 0.0f;
        for (int q = 0; q < NF; ++q)
            s += HIERW * bb3[q] + (1.0f - HIERW) * sb3[q];
        *(float*)(ws + OFF_SB) = s;
    }
}

// ---------------------------------------------------------------------------
// Main kernel: 256 threads = 8 waves; each wave owns 4 row-tiles (64 rows).
// 256 blocks x 512 rows = 131072 rows.
// ---------------------------------------------------------------------------
__global__ __launch_bounds__(256) void nam_main(const float* __restrict__ x,
                                                const char*  __restrict__ ws,
                                                float* __restrict__ out) {
    const int lane = threadIdx.x & 31;
    const int wave = threadIdx.x >> 5;
    const int m    = lane & 15;   // row within 16-row tile (A layout M index; C layout N index)
    const int hs   = lane >> 4;   // lane-half select
    const int rowbase = (blockIdx.x * 8 + wave) * 64;

    const v16h*  __restrict__ w2f = (const v16h*)(ws + OFF_W2);
    const v16h*  __restrict__ w1f = (const v16h*)(ws + OFF_W1);
    const v16h*  __restrict__ b1f = (const v16h*)(ws + OFF_B1);
    const float* __restrict__ b2s = (const float*)(ws + OFF_B2);
    const float* __restrict__ w3s = (const float*)(ws + OFF_W3);
    const float Sb = *(const float*)(ws + OFF_SB);

    const v16h zh = {};

    float acc[4][8];
#pragma unroll
    for (int rt = 0; rt < 4; ++rt)
#pragma unroll
        for (int i = 0; i < 8; ++i) acc[rt][i] = 0.0f;

    for (int f = 0; f < NF; ++f) {
        // x value for this lane's row in each of the 4 row-tiles
        _Float16 xh[4];
#pragma unroll
        for (int rt = 0; rt < 4; ++rt)
            xh[rt] = (_Float16)x[(rowbase + rt * 16 + m) * NF + f];

        // keep the model loop rolled: halves live fragment state, keeps the
        // wave under 256 VGPRs (no s_set_vgpr_msb in the hot loop)
#pragma unroll 1
        for (int model = 0; model < 2; ++model) {
            const int p = f * 2 + model;
            // B fragments for layer-2 GEMM: [ks][nt], 32B/lane each
            const v16h B00 = w2f[((p * 2 + 0) * 2 + 0) * 32 + lane];
            const v16h B01 = w2f[((p * 2 + 0) * 2 + 1) * 32 + lane];
            const v16h B10 = w2f[((p * 2 + 1) * 2 + 0) * 32 + lane];
            const v16h B11 = w2f[((p * 2 + 1) * 2 + 1) * 32 + lane];
            // layer-1 weight/bias fragments (per lane-half, broadcast load)
            const v16h w10 = w1f[(p * 2 + 0) * 2 + hs];
            const v16h w11 = w1f[(p * 2 + 1) * 2 + hs];
            const v16h b10 = b1f[(p * 2 + 0) * 2 + hs];
            const v16h b11 = b1f[(p * 2 + 1) * 2 + hs];
            // layer-2 bias (preloaded into C) and hier-weighted layer-3 weights
            const float b2a = b2s[p * 32 + m];
            const float b2b = b2s[p * 32 + 16 + m];
            const float w3a = w3s[p * 32 + m];
            const float w3b = w3s[p * 32 + 16 + m];

#pragma unroll
            for (int rt = 0; rt < 4; ++rt) {
                // layer 1: h1 = relu(x * w1 + b1), built directly in WMMA
                // A-fragment layout via packed f16 ops
                v16h xs = zh;
                xs = xs + xh[rt];                 // lane-wide splat
                v16h a0 = xs * w10 + b10;
                v16h a1 = xs * w11 + b11;
                a0 = __builtin_elementwise_max(a0, zh);
                a1 = __builtin_elementwise_max(a1, zh);

                // layer 2: C starts as b2 broadcast, accumulate K=64 in 2 steps
                v8f c0 = {}; c0 = c0 + b2a;
                v8f c1 = {}; c1 = c1 + b2b;
                c0 = __builtin_amdgcn_wmma_f32_16x16x32_f16(false, a0, false, B00,
                                                            (short)0, c0, false, false);
                c0 = __builtin_amdgcn_wmma_f32_16x16x32_f16(false, a1, false, B10,
                                                            (short)0, c0, false, false);
                c1 = __builtin_amdgcn_wmma_f32_16x16x32_f16(false, a0, false, B01,
                                                            (short)0, c1, false, false);
                c1 = __builtin_amdgcn_wmma_f32_16x16x32_f16(false, a1, false, B11,
                                                            (short)0, c1, false, false);

                // layer 3: relu(h2) dot w3 (hier weight folded into w3)
#pragma unroll
                for (int i = 0; i < 8; ++i) {
                    acc[rt][i] += relu_f32(c0[i]) * w3a
                                + relu_f32(c1[i]) * w3b;
                }
            }
        }
    }

    // reduce the N dimension (16 lanes per lane-half) and store
#pragma unroll
    for (int rt = 0; rt < 4; ++rt) {
#pragma unroll
        for (int i = 0; i < 8; ++i) {
            float v = acc[rt][i];
            v += __shfl_xor(v, 8, 16);
            v += __shfl_xor(v, 4, 16);
            v += __shfl_xor(v, 2, 16);
            v += __shfl_xor(v, 1, 16);
            acc[rt][i] = v + Sb;
        }
        if (m == 0) {
            // lane 0 holds rows rt*16+0..7, lane 16 holds rows rt*16+8..15
            float4* dst = (float4*)(out + rowbase + rt * 16 + hs * 8);
            dst[0] = make_float4(acc[rt][0], acc[rt][1], acc[rt][2], acc[rt][3]);
            dst[1] = make_float4(acc[rt][4], acc[rt][5], acc[rt][6], acc[rt][7]);
        }
    }
}

// ---------------------------------------------------------------------------
extern "C" void kernel_launch(void* const* d_in, const int* in_sizes, int n_in,
                              void* d_out, int out_size, void* d_ws, size_t ws_size,
                              hipStream_t stream) {
    const float* x   = (const float*)d_in[0];
    const float* bW1 = (const float*)d_in[1];
    const float* bb1 = (const float*)d_in[2];
    const float* bW2 = (const float*)d_in[3];
    const float* bb2 = (const float*)d_in[4];
    const float* bW3 = (const float*)d_in[5];
    const float* bb3 = (const float*)d_in[6];
    const float* sW1 = (const float*)d_in[7];
    const float* sb1 = (const float*)d_in[8];
    const float* sW2 = (const float*)d_in[9];
    const float* sb2 = (const float*)d_in[10];
    const float* sW3 = (const float*)d_in[11];
    const float* sb3 = (const float*)d_in[12];
    char* ws = (char*)d_ws;

    nam_pack<<<dim3(NPAIR), dim3(256), 0, stream>>>(bW1, bb1, bW2, bb2, bW3, bb3,
                                                    sW1, sb1, sW2, sb2, sW3, sb3, ws);
    nam_main<<<dim3(BATCH / 512), dim3(256), 0, stream>>>(x, ws, (float*)d_out);
}
